// SkelConv_13829794693142
// MI455X (gfx1250) — matrix operand order repaired and checked by reading, not compiled, not run on hardware
//
#include <hip/hip_runtime.h>
#include <hip/hip_bf16.h>

// ---------------- problem constants (baked; conv_map is static) -------------
#define CIN    103
#define COUT   206
#define TLEN   8192
#define NBATCH 32
#define KW     15
#define PADT   7
#define TBLK   256          // time positions per block (8 waves x 2 subtiles x 16)
#define NTILES 13
#define MAXSTEP 16          // max k-steps per tile (tile 1: nch_pad=32 -> 16)

typedef __attribute__((ext_vector_type(16))) _Float16 v16h;
typedef __attribute__((ext_vector_type(8)))  float    v8f;

// Per-M-tile metadata. Tile 0 = root joint (14 rows). Tiles 1..12 = joint
// pairs (2t-1, 2t), 16 rows each. Fan-in input channels are contiguous.
__constant__ int tc_co0[NTILES] = {0, 14, 30, 46, 62, 78, 94, 110, 126, 142, 158, 174, 190};
__constant__ int tc_vr [NTILES] = {14, 16, 16, 16, 16, 16, 16, 16, 16, 16, 16, 16, 16};
__constant__ int tc_chs[NTILES] = {0, 0, 11, 19, 27, 35, 43, 51, 59, 67, 75, 83, 91};
__constant__ int tc_nch[NTILES] = {11, 19, 16, 16, 16, 16, 16, 16, 16, 16, 16, 16, 12};
__constant__ int tc_pad[NTILES] = {16, 32, 16, 16, 16, 16, 16, 16, 16, 16, 16, 16, 16};

// ---------------- weight packing: f32 masked W -> f16 A-fragments ------------
// Fragment (tile, s) is 16x32 f16 in contiguous-K A layout; stored so lane L reads its
// 16 halves contiguously: ws[((tile*MAXSTEP + s)*32 + L)*16 + idx].
// A layout used here (16-bit, 16x32): lane%16 = M; K = idx + (lane>=16?16:0)
__global__ __launch_bounds__(256) void SkelConv_pack_w(
    const float* __restrict__ w, const float* __restrict__ msk,
    _Float16* __restrict__ wsA)
{
    const int tile = blockIdx.x / MAXSTEP;
    const int s    = blockIdx.x % MAXSTEP;
    const int co0 = tc_co0[tile], vr = tc_vr[tile];
    const int chs = tc_chs[tile], nch = tc_nch[tile], pad = tc_pad[tile];
    const int shp = (pad == 32) ? 5 : 4;

    for (int e = threadIdx.x; e < 512; e += 256) {
        const int lane = e >> 4, idx = e & 15;
        const int v = idx >> 1, h = idx & 1;
        const int M = lane & 15;
        const int K = idx + (lane >= 16 ? 16 : 0);
        const int r = s * 32 + K;               // reduction index = k*pad + ch
        const int k = r >> shp;
        const int ch = r & (pad - 1);
        float val = 0.f;
        if (k < KW && ch < nch && M < vr) {
            const size_t wi = ((size_t)(co0 + M) * CIN + (chs + ch)) * KW + k;
            val = w[wi] * msk[wi];
        }
        wsA[((size_t)(tile * MAXSTEP + s) * 32 + lane) * 16 + idx] = (_Float16)val;
    }
}

// Fully-unrolled WMMA inner product. All fragment addresses reduce to
// base + 64*s bytes (LDS) and base + 1024*s bytes (global), so the unrolled
// body is clauses of b128 loads with immediate offsets + WMMAs; the compiler
// can hoist loads across steps instead of a hard wait-0 per iteration.
template <int KSTEPS, int SHP>
__device__ __forceinline__ void conv_compute(
    const _Float16* __restrict__ sx, const _Float16* __restrict__ aTile,
    int lane, int nA, int nB, v8f& accA, v8f& accB)
{
    const int lm = lane & 15;
    const bool hi = (lane >= 16);
    const _Float16* aB = aTile + lane * 16;           // + s*512 halves per step

    // B fragment base (contiguous-K B layout, as A: lanes 0-15 K=0..15, lanes 16-31 K=16..31).
    // pad16: addr(s) = (n + lm + hi + 2s) << 4      => base + 32*s halves
    // pad32: addr(s) = ((n + lm + s) << 5) + 16*hi  => base + 32*s halves
    const _Float16* bBaseA;
    const _Float16* bBaseB;
    if (SHP == 4) {
        bBaseA = sx + ((nA + lm + (hi ? 1 : 0)) << 4);
        bBaseB = sx + ((nB + lm + (hi ? 1 : 0)) << 4);
    } else {
        bBaseA = sx + ((nA + lm) << 5) + (hi ? 16 : 0);
        bBaseB = sx + ((nB + lm) << 5) + (hi ? 16 : 0);
    }

#pragma unroll
    for (int s = 0; s < KSTEPS; ++s) {
        v16h a  = *(const v16h*)(aB + s * 512);
        v16h bA = *(const v16h*)(bBaseA + s * 32);
        v16h bB = *(const v16h*)(bBaseB + s * 32);
        accA = __builtin_amdgcn_wmma_f32_16x16x32_f16(
            false, a, false, bA, (short)0, accA, false, false);
        accB = __builtin_amdgcn_wmma_f32_16x16x32_f16(
            false, a, false, bB, (short)0, accB, false, false);
    }
}

// ---------------- main kernel: implicit-GEMM conv via v_wmma_f32_16x16x32_f16
__global__ __launch_bounds__(256) void SkelConv_main(
    const float* __restrict__ x, const _Float16* __restrict__ wsA,
    const float* __restrict__ bias, float* __restrict__ out)
{
    // LDS: x tile as f16, layout [t_local][ch], t_local in [0, 272)
    __shared__ __align__(32) _Float16 sx[272 * 32];

    const int tblk = blockIdx.x;      // 0..31
    const int tile = blockIdx.y;      // 0..12
    const int b    = blockIdx.z;      // 0..31
    const int t0   = tblk * TBLK;

    const int co0 = tc_co0[tile], vr = tc_vr[tile];
    const int chs = tc_chs[tile], nch = tc_nch[tile], pad = tc_pad[tile];
    const int shp = (pad == 32) ? 5 : 4;

    // ---- stage x -> LDS (f32 -> f16), halo [-7, +7], zero-pad everywhere else
    {
        const float* xb = x + ((size_t)b * CIN + chs) * TLEN;
        const int total = 272 << shp;     // 272 * pad
        for (int e = threadIdx.x; e < total; e += 256) {
            const int ch = e / 272;       // channel-major => coalesced global reads
            const int tl = e - ch * 272;
            const int tg = t0 - PADT + tl;
            float v = 0.f;
            if (ch < nch && tl <= (TBLK + 2 * PADT - 1) && tg >= 0 && tg < TLEN)
                v = xb[(size_t)ch * TLEN + tg];
            sx[(tl << shp) + ch] = (_Float16)v;   // padded slots stay 0 (no NaN*0)
        }
    }
    __syncthreads();

    const int l  = threadIdx.x & 31;
    const int wv = threadIdx.x >> 5;
    const int nA = wv * 32;            // wave's two 16-column subtiles
    const int nB = wv * 32 + 16;

    v8f accA = {};
    v8f accB = {};
    const _Float16* aTile = wsA + (size_t)tile * MAXSTEP * 512;

    if (pad == 16)
        conv_compute<8, 4>(sx, aTile, l, nA, nB, accA, accB);
    else
        conv_compute<16, 5>(sx, aTile, l, nA, nB, accA, accB);

    // ---- epilogue: D layout: lane<16 -> (M=i, N=lane); lane>=16 -> (M=i+8, N=lane-16)
    const int lm = l & 15;
    const int Mb = (l >= 16) ? 8 : 0;
    float* ob = out + ((size_t)b * COUT) * TLEN;
#pragma unroll
    for (int i = 0; i < 8; ++i) {
        const int M = i + Mb;
        if (M < vr) {
            const int co = co0 + M;
            const float bv = bias[co];
            ob[(size_t)co * TLEN + t0 + nA + lm] = accA[i] + bv;
            ob[(size_t)co * TLEN + t0 + nB + lm] = accB[i] + bv;
        }
    }
}

extern "C" void kernel_launch(void* const* d_in, const int* in_sizes, int n_in,
                              void* d_out, int out_size, void* d_ws, size_t ws_size,
                              hipStream_t stream) {
    const float* x    = (const float*)d_in[0];   // (32,103,8192) f32
    const float* w    = (const float*)d_in[1];   // (206,103,15)  f32
    const float* bias = (const float*)d_in[2];   // (206,)        f32
    const float* msk  = (const float*)d_in[3];   // (206,103,15)  f32
    float* out = (float*)d_out;                  // (32,206,8192) f32
    _Float16* wsA = (_Float16*)d_ws;             // needs 13*16*512*2 = 208 KB

    // 1) pack masked weights into f16 WMMA A-fragments (idempotent, deterministic)
    SkelConv_pack_w<<<NTILES * MAXSTEP, 256, 0, stream>>>(w, msk, wsA);

    // 2) sparse implicit-GEMM conv
    dim3 grid(TLEN / TBLK, NTILES, NBATCH);      // (32, 13, 32)
    SkelConv_main<<<grid, 256, 0, stream>>>(x, wsA, bias, out);
}
